// GraphReasonLayer_89361089561155
// MI455X (gfx1250) — matrix operand order repaired and checked by reading, not compiled, not run on hardware
//
#include <hip/hip_runtime.h>
#include <hip/hip_bf16.h>

// ---------------------------------------------------------------------------
// GraphReasonLayer for MI455X (gfx1250): bf16 WMMA pipeline, 8 waves/WG
//   0) prep: W^T -> bf16 (x3), inp -> bf16 (row-major + per-batch transpose)
//   1) Q = inp@Wq+bq, K = inp@Wk+bk             (WMMA, bf16 out to ws)
//   2) scores = Q@K^T -> softmax -> agg = P@inp (WMMA + LDS softmax)
//   3) out = relu(agg@Wo+bo)                    (WMMA, f32 out)
// Every WMMA operand in a hot loop is a contiguous bf16 vector load (b128);
// all f32->bf16 conversion happens once in the streaming prep kernels.
// ---------------------------------------------------------------------------

typedef __attribute__((ext_vector_type(16))) __bf16 v16bf;
typedef __attribute__((ext_vector_type(8)))  __bf16 v8bf;
typedef __attribute__((ext_vector_type(4)))  __bf16 v4bf;
typedef __attribute__((ext_vector_type(8)))  float  v8f;
typedef __bf16 bf16_t;

constexpr int B_ = 16;
constexpr int N_ = 2048;
constexpr int D_ = 256;

// --- WMMA wrapper -----------------------------------------------------------
__device__ __forceinline__ v8f wmma_bf16(v16bf a, v16bf b, v8f c) {
  return __builtin_amdgcn_wmma_f32_16x16x32_bf16(
      /*neg_a=*/false, a, /*neg_b=*/false, b,
      /*c_mod=*/(short)0, c, /*reuse_a=*/false, /*reuse_b=*/false);
}

// A (16x32) fragment from row-major bf16 [ld] (global or LDS).
// Lane data = two contiguous 8-elem chunks at k0+16h and k0+16h+8 -> 2x b128.
__device__ __forceinline__ v16bf load_a_bf16(const bf16_t* __restrict__ src,
                                             int ld, int row0, int k0) {
  const int lane = threadIdx.x & 31;
  const int m = lane & 15, h = lane >> 4;
  const bf16_t* p = src + (size_t)(row0 + m) * ld + k0 + 16 * h;
  v8bf lo = *(const v8bf*)(p);
  v8bf hi = *(const v8bf*)(p + 8);
  return __builtin_shufflevector(lo, hi, 0, 1, 2, 3, 4, 5, 6, 7,
                                 8, 9, 10, 11, 12, 13, 14, 15);
}

// B (32x16) fragment where B[k][n] = srcT[n0+n][k0+k] (B = srcT^T),
// srcT row-major bf16 [ld]. Lane data = 16 contiguous bf16 -> 32B load.
__device__ __forceinline__ v16bf load_bT_bf16(const bf16_t* __restrict__ srcT,
                                              int ld, int k0, int n0) {
  const int lane = threadIdx.x & 31;
  const int n = lane & 15, h = lane >> 4;
  const bf16_t* p = srcT + (size_t)(n0 + n) * ld + k0 + 16 * h;
  return *(const v16bf*)p;
}

// ---------------------------------------------------------------------------
// Prep A: transpose+convert weights -> bf16 W^T[out][in] (256x256 each)
// ---------------------------------------------------------------------------
__global__ void prep_weights(const float* __restrict__ Wq,
                             const float* __restrict__ Wk,
                             const float* __restrict__ Wo,
                             bf16_t* __restrict__ WqT,
                             bf16_t* __restrict__ WkT,
                             bf16_t* __restrict__ WoT) {
  const int o = blockIdx.x;    // 0..255 (output col of W)
  const int i = threadIdx.x;   // 0..255 (input row of W)
  WqT[(size_t)o * D_ + i] = (bf16_t)Wq[(size_t)i * D_ + o];
  WkT[(size_t)o * D_ + i] = (bf16_t)Wk[(size_t)i * D_ + o];
  WoT[(size_t)o * D_ + i] = (bf16_t)Wo[(size_t)i * D_ + o];
}

// ---------------------------------------------------------------------------
// Prep B: inp f32 -> bf16 row-major copy AND per-batch transpose (LDS tiles)
//   inpb[b*N+n][d]  = bf16(inp[b][n][d])   (coalesced)
//   inpT[b][d][n]   = bf16(inp[b][n][d])
// ---------------------------------------------------------------------------
__global__ void prep_inp(const float* __restrict__ inp,
                         bf16_t* __restrict__ inpb,
                         bf16_t* __restrict__ inpT) {
  __shared__ bf16_t tile[32][33];
  const int bt = blockIdx.z;
  const int d0 = blockIdx.x * 32, n0 = blockIdx.y * 32;
  const float* src = inp + (size_t)bt * N_ * D_;
  bf16_t* dstR = inpb + (size_t)bt * N_ * D_;
  bf16_t* dstT = inpT + (size_t)bt * D_ * N_;
#pragma unroll
  for (int r = 0; r < 32; r += 8) {
    const int n = n0 + threadIdx.y + r;
    const int d = d0 + threadIdx.x;
    const bf16_t v = (bf16_t)src[(size_t)n * D_ + d];
    tile[threadIdx.y + r][threadIdx.x] = v;
    dstR[(size_t)n * D_ + d] = v;                 // coalesced row-major copy
  }
  __syncthreads();
#pragma unroll
  for (int r = 0; r < 32; r += 8) {
    const int d = d0 + threadIdx.y + r;
    const int n = n0 + threadIdx.x;
    dstT[(size_t)d * N_ + n] = tile[threadIdx.x][threadIdx.y + r];
  }
}

// ---------------------------------------------------------------------------
// Kernel 1: Q/K projections. One 16-row tile (over flattened B*N) per WG.
// 8 waves of 32; wave w owns output columns [32w, 32w+32).
// ---------------------------------------------------------------------------
__global__ void proj_qk_kernel(const bf16_t* __restrict__ inpb,
                               const bf16_t* __restrict__ WqT,
                               const float* __restrict__ bq,
                               const bf16_t* __restrict__ WkT,
                               const float* __restrict__ bk,
                               bf16_t* __restrict__ Qb,
                               bf16_t* __restrict__ Kb) {
  const int R = blockIdx.x * 16;
  const int wave = threadIdx.x >> 5;    // 0..7
  const int lane = threadIdx.x & 31;
  const int n16 = lane & 15, h = lane >> 4;

  v16bf af[8];
#pragma unroll
  for (int kk = 0; kk < 8; ++kk) af[kk] = load_a_bf16(inpb, D_, R, kk * 32);

#pragma unroll
  for (int which = 0; which < 2; ++which) {
    const bf16_t* WT = which ? WkT : WqT;
    const float* bias = which ? bk : bq;
    bf16_t* out = which ? Kb : Qb;
#pragma unroll
    for (int ct = 0; ct < 2; ++ct) {
      const int col0 = wave * 32 + ct * 16;
      v8f acc = {};
#pragma unroll
      for (int kk = 0; kk < 8; ++kk)
        acc = wmma_bf16(af[kk], load_bT_bf16(WT, D_, kk * 32, col0), acc);
      const float bv = bias[col0 + n16];
#pragma unroll
      for (int r = 0; r < 8; ++r)
        out[(size_t)(R + r + 8 * h) * D_ + col0 + n16] = (bf16_t)(acc[r] + bv);
    }
  }
}

// ---------------------------------------------------------------------------
// Kernel 2: attention. One (batch, 16-row tile) per WG of 256 threads.
// LDS: sc[16][N] f32 scores (128KB) + pb[16][N] bf16 probs (64KB) + partials.
// ---------------------------------------------------------------------------
__global__ void attn_kernel(const bf16_t* __restrict__ inpT,
                            const bf16_t* __restrict__ Qb,
                            const bf16_t* __restrict__ Kb,
                            bf16_t* __restrict__ aggb) {
  const int bt = blockIdx.x >> 7;   // / (N/16)
  const int nt = blockIdx.x & 127;
  const int wave = threadIdx.x >> 5;   // 0..7
  const int lane = threadIdx.x & 31;
  const int n16 = lane & 15, h = lane >> 4;

  extern __shared__ char smem[];
  float* sc = (float*)smem;                                    // 16*N f32
  bf16_t* pb = (bf16_t*)(smem + (size_t)16 * N_ * 4);          // 16*N bf16
  float* red = (float*)(smem + (size_t)16 * N_ * 6);           // 16*16 f32
  float* red2 = red + 16 * 16;                                 // 16*16 f32

  const bf16_t* Qrow = Qb + ((size_t)bt * N_ + (size_t)nt * 16) * D_;
  const bf16_t* Kbase = Kb + (size_t)bt * N_ * D_;
  const bf16_t* inpTB = inpT + (size_t)bt * D_ * N_;

  // ---- phase 1: scores = Q @ K^T into LDS (m-tiles strided over 8 waves) ---
  v16bf af[8];
#pragma unroll
  for (int kk = 0; kk < 8; ++kk) af[kk] = load_a_bf16(Qrow, D_, 0, kk * 32);

  for (int mt = wave; mt < N_ / 16; mt += 8) {
    v8f s = {};
#pragma unroll
    for (int kk = 0; kk < 8; ++kk)
      s = wmma_bf16(af[kk], load_bT_bf16(Kbase, D_, kk * 32, mt * 16), s);
#pragma unroll
    for (int r = 0; r < 8; ++r)
      sc[(r + 8 * h) * N_ + mt * 16 + n16] = s[r];
  }
  __syncthreads();

  // ---- phase 2: row softmax (16 rows x 16 threads/row), float4 vectorized --
  {
    const int row = threadIdx.x & 15;
    const int chunk = threadIdx.x >> 4;   // 0..15
    const int span = N_ / 16;             // 128
    float4* rp4 = (float4*)(sc + row * N_ + chunk * span);

    float lmax = -3.4e38f;
    for (int i = 0; i < span / 4; ++i) {
      float4 v = rp4[i];
      lmax = fmaxf(lmax, fmaxf(fmaxf(v.x, v.y), fmaxf(v.z, v.w)));
    }
    red[row * 16 + chunk] = lmax;
    __syncthreads();

    float m = red[row * 16];
#pragma unroll
    for (int c = 1; c < 16; ++c) m = fmaxf(m, red[row * 16 + c]);

    float lsum = 0.f;
    for (int i = 0; i < span / 4; ++i) {
      float4 v = rp4[i];
      v.x = __expf(v.x - m);
      v.y = __expf(v.y - m);
      v.z = __expf(v.z - m);
      v.w = __expf(v.w - m);
      rp4[i] = v;
      lsum += (v.x + v.y) + (v.z + v.w);
    }
    red2[row * 16 + chunk] = lsum;
    __syncthreads();

    float ssum = 0.f;
#pragma unroll
    for (int c = 0; c < 16; ++c) ssum += red2[row * 16 + c];
    const float inv = 1.0f / ssum;

    v4bf* pp4 = (v4bf*)(pb + row * N_ + chunk * span);
    for (int i = 0; i < span / 4; ++i) {
      float4 v = rp4[i];
      v4bf w;
      w[0] = (bf16_t)(v.x * inv);
      w[1] = (bf16_t)(v.y * inv);
      w[2] = (bf16_t)(v.z * inv);
      w[3] = (bf16_t)(v.w * inv);
      pp4[i] = w;
    }
  }
  __syncthreads();

  // ---- phase 3: agg = P @ inp (K-dim = N). Wave w owns cols [32w,32w+32) ---
  // A from LDS (2x ds b128, prefetched one iteration ahead); B from inpT
  // (contiguous 32B per lane). Unroll by 2 so loads interleave with WMMAs.
  const int c0 = wave * 32;
  v8f acc0 = {}, acc1 = {};
  v16bf a = load_a_bf16(pb, N_, 0, 0);
#pragma unroll 2
  for (int kk = 0; kk < N_ / 32; ++kk) {
    v16bf b0 = load_bT_bf16(inpTB, N_, kk * 32, c0);
    v16bf b1 = load_bT_bf16(inpTB, N_, kk * 32, c0 + 16);
    v16bf an = (kk + 1 < N_ / 32) ? load_a_bf16(pb, N_, 0, (kk + 1) * 32) : a;
    acc0 = wmma_bf16(a, b0, acc0);
    acc1 = wmma_bf16(a, b1, acc1);
    a = an;
  }
#pragma unroll
  for (int r = 0; r < 8; ++r) {
    const size_t rowAddr = ((size_t)bt * N_ + nt * 16 + r + 8 * h) * D_;
    aggb[rowAddr + c0 + n16] = (bf16_t)acc0[r];
    aggb[rowAddr + c0 + 16 + n16] = (bf16_t)acc1[r];
  }
}

// ---------------------------------------------------------------------------
// Kernel 3: out = relu(agg @ Wo + bo), f32 output. 8 waves, 2 col-tiles each.
// ---------------------------------------------------------------------------
__global__ void out_proj_kernel(const bf16_t* __restrict__ aggb,
                                const bf16_t* __restrict__ WoT,
                                const float* __restrict__ bo,
                                float* __restrict__ out) {
  const int R = blockIdx.x * 16;
  const int wave = threadIdx.x >> 5;   // 0..7
  const int lane = threadIdx.x & 31;
  const int n16 = lane & 15, h = lane >> 4;

  v16bf af[8];
#pragma unroll
  for (int kk = 0; kk < 8; ++kk) af[kk] = load_a_bf16(aggb, D_, R, kk * 32);

#pragma unroll
  for (int ct = 0; ct < 2; ++ct) {
    const int col0 = wave * 32 + ct * 16;
    v8f acc = {};
#pragma unroll
    for (int kk = 0; kk < 8; ++kk)
      acc = wmma_bf16(af[kk], load_bT_bf16(WoT, D_, kk * 32, col0), acc);
    const float bv = bo[col0 + n16];
#pragma unroll
    for (int r = 0; r < 8; ++r)
      out[(size_t)(R + r + 8 * h) * D_ + col0 + n16] =
          fmaxf(acc[r] + bv, 0.0f);
  }
}

// ---------------------------------------------------------------------------
extern "C" void kernel_launch(void* const* d_in, const int* in_sizes, int n_in,
                              void* d_out, int out_size, void* d_ws,
                              size_t ws_size, hipStream_t stream) {
  const float* inp = (const float*)d_in[0];
  const float* Wq = (const float*)d_in[1];
  const float* bq = (const float*)d_in[2];
  const float* Wk = (const float*)d_in[3];
  const float* bk = (const float*)d_in[4];
  const float* Wo = (const float*)d_in[5];
  const float* bo = (const float*)d_in[6];
  float* out = (float*)d_out;

  const size_t elems = (size_t)B_ * N_ * D_;   // 8M
  const size_t welems = (size_t)D_ * D_;       // 64K
  bf16_t* Qb = (bf16_t*)d_ws;
  bf16_t* Kb = Qb + elems;
  bf16_t* aggb = Kb + elems;
  bf16_t* inpT = aggb + elems;
  bf16_t* inpb = inpT + elems;
  bf16_t* WqT = inpb + elems;
  bf16_t* WkT = WqT + welems;
  bf16_t* WoT = WkT + welems;

  // prep
  prep_weights<<<D_, D_, 0, stream>>>(Wq, Wk, Wo, WqT, WkT, WoT);
  prep_inp<<<dim3(D_ / 32, N_ / 32, B_), dim3(32, 8), 0, stream>>>(inp, inpb,
                                                                   inpT);

  const int rowTiles = B_ * N_ / 16;  // 2048
  proj_qk_kernel<<<rowTiles, 256, 0, stream>>>(inpb, WqT, bq, WkT, bk, Qb, Kb);

  const int smemBytes =
      16 * N_ * 4 + 16 * N_ * 2 + 2 * 16 * 16 * (int)sizeof(float);
  (void)hipFuncSetAttribute(reinterpret_cast<const void*>(attn_kernel),
                            hipFuncAttributeMaxDynamicSharedMemorySize,
                            smemBytes);
  attn_kernel<<<B_ * (N_ / 16), 256, smemBytes, stream>>>(inpT, Qb, Kb, aggb);

  out_proj_kernel<<<rowTiles, 256, 0, stream>>>(aggb, WoT, bo, out);
}
